// LSTMSubTrajectoryBasedSkillGenerator_32925219291296
// MI455X (gfx1250) — compile-verified, matrix operand order checked by reading, not run on hardware
//
#include <hip/hip_runtime.h>

// ---------------------------------------------------------------------------
// LSTM skill-generator for MI455X (gfx1250, wave32, WMMA bf16).
//   B=1024, T=64, OBS=256, ACT=16, EMB=128, H=128, OUT=64
// Pipeline:
//   prep_weights : fp32 -> bf16 transpose of all weight matrices (B-fragment rows)
//   embed_kernel : emb = concat(obs,act) @ We + b, written pre-packed as WMMA
//                  A-fragments [btile][t][ksub][lane][16 bf16]
//   lstm_kernel  : 64 sequential steps; per-wave Wi/Wh fragments live in
//                  REGISTERS; x_t double-buffered in LDS via
//                  global_load_async_to_lds_b128 (ASYNCcnt); fused head + clip.
// ---------------------------------------------------------------------------

#define BB   1024
#define TT   64
#define NOBS 256
#define NACT 16
#define NEMB 128
#define NH   128
#define NOUT 64

typedef __bf16 v16bf __attribute__((ext_vector_type(16)));
typedef float  v8f   __attribute__((ext_vector_type(8)));

union V16BF { unsigned int u[8]; uint4 q[2]; v16bf v; };

static __device__ __forceinline__ v16bf make_v16bf(uint4 lo, uint4 hi) {
  V16BF t; t.q[0] = lo; t.q[1] = hi;
  return t.v;
}

static __device__ __forceinline__ float sigmoidf_(float x) {
  return 1.0f / (1.0f + __expf(-x));
}

// Async-copy 4096 B (one packed x_t block) global -> LDS using one wave.
// Per lane: base + lane*16; 8 issues at 512 B immediate offsets.
static __device__ __forceinline__ void async_copy_x4k(unsigned ldsBase,
                                                      const char* gsrc,
                                                      int lane) {
  unsigned lo = ldsBase + lane * 16;
  unsigned long long ga = (unsigned long long)(const void*)(gsrc + lane * 16);
  asm volatile(
      "global_load_async_to_lds_b128 %0, %1, off\n\t"
      "global_load_async_to_lds_b128 %0, %1, off offset:512\n\t"
      "global_load_async_to_lds_b128 %0, %1, off offset:1024\n\t"
      "global_load_async_to_lds_b128 %0, %1, off offset:1536\n\t"
      "global_load_async_to_lds_b128 %0, %1, off offset:2048\n\t"
      "global_load_async_to_lds_b128 %0, %1, off offset:2560\n\t"
      "global_load_async_to_lds_b128 %0, %1, off offset:3072\n\t"
      "global_load_async_to_lds_b128 %0, %1, off offset:3584"
      :: "v"(lo), "v"(ga) : "memory");
}

static __device__ __forceinline__ void wait_async0() {
  asm volatile("s_wait_asynccnt 0x0" ::: "memory");
}

// ---------------------------------------------------------------------------
// Kernel 0: weight prep. Transposed bf16 rows: Wt[n][k] = bf16(W[k][n]).
// WtE is [128][288] with rows k>=272 zero-padded.
// ---------------------------------------------------------------------------
__global__ void prep_weights(const float* __restrict__ eW,   // [272][128]
                             const float* __restrict__ Wi,   // [128][512]
                             const float* __restrict__ Wh,   // [128][512]
                             const float* __restrict__ muW,  // [128][128]
                             __bf16* __restrict__ WtE,
                             __bf16* __restrict__ WtI,
                             __bf16* __restrict__ WtH,
                             __bf16* __restrict__ WtMu) {
  int idx = blockIdx.x * blockDim.x + threadIdx.x;
  const int nE = 128 * 288;
  const int nI = 512 * 128;
  const int nM = 128 * 128;
  if (idx < nE) {
    int n = idx / 288, k = idx % 288;
    float v = (k < (NOBS + NACT)) ? eW[k * NEMB + n] : 0.0f;
    WtE[idx] = (__bf16)v;
  } else if (idx < nE + nI) {
    int i = idx - nE; int n = i / 128, k = i % 128;
    WtI[i] = (__bf16)Wi[k * 512 + n];
  } else if (idx < nE + 2 * nI) {
    int i = idx - nE - nI; int n = i / 128, k = i % 128;
    WtH[i] = (__bf16)Wh[k * 512 + n];
  } else if (idx < nE + 2 * nI + nM) {
    int i = idx - nE - 2 * nI; int n = i / 128, k = i % 128;
    WtMu[i] = (__bf16)muW[k * 128 + n];
  }
}

// ---------------------------------------------------------------------------
// Kernel 1: embed GEMM. grid = (btile, t), 128 threads (4 waves).
// Each block: 16 batch rows at one timestep, N=128, K=272 (8 obs + 1 act step).
// Output packed as WMMA A-fragments for the LSTM.
// ---------------------------------------------------------------------------
__global__ __launch_bounds__(128)
void embed_kernel(const float* __restrict__ obs,
                  const float* __restrict__ act,
                  const __bf16* __restrict__ WtE,           // [128][288] bf16
                  const float* __restrict__ eb,             // [128]
                  __bf16* __restrict__ embP) {              // packed A-fragments
  __shared__ float estage[16 * 132];

  const int btile = blockIdx.x;       // 0..63 (16 batch rows each)
  const int t     = blockIdx.y;       // 0..63
  const int tid   = threadIdx.x;
  const int lane  = tid & 31;
  const int wid   = tid >> 5;         // 0..3
  const int m     = lane & 15;
  const int half  = lane >> 4;
  const int nbase = wid * 32;

  const long grow = (long)(btile * 16 + m) * TT + t;   // flattened (b,t) row
  const float* obsrow = obs + grow * NOBS;
  const float* actrow = act + grow * NACT;

  v8f acc0 = {}; v8f acc1 = {};

  #pragma unroll
  for (int ks = 0; ks < 9; ++ks) {
    // ---- A fragment: fp32 activations -> bf16 (native cvt) ----------------
    float fa[16];
    if (ks < 8) {
      const int k0 = 32 * ks + 8 * half;
      #pragma unroll
      for (int i = 0; i < 8; ++i) {
        fa[i]     = obsrow[k0 + i];
        fa[8 + i] = obsrow[k0 + 16 + i];
      }
    } else {                       // K = 256..287: act columns + zero pad
      const int k0 = 8 * half;
      #pragma unroll
      for (int i = 0; i < 8; ++i) {
        fa[i]     = actrow[k0 + i];
        fa[8 + i] = 0.0f;
      }
    }
    v16bf A;
    #pragma unroll
    for (int i = 0; i < 16; ++i) A[i] = (__bf16)fa[i];

    // ---- B fragments from transposed weights ------------------------------
    const __bf16* w0 = WtE + (size_t)(nbase + m) * 288 + 32 * ks + 8 * half;
    const __bf16* w1 = w0 + (size_t)16 * 288;
    v16bf B0 = make_v16bf(*(const uint4*)w0, *(const uint4*)(w0 + 16));
    v16bf B1 = make_v16bf(*(const uint4*)w1, *(const uint4*)(w1 + 16));

    acc0 = __builtin_amdgcn_wmma_f32_16x16x32_bf16(false, A, false, B0, (short)0, acc0, false, false);
    acc1 = __builtin_amdgcn_wmma_f32_16x16x32_bf16(false, A, false, B1, (short)0, acc1, false, false);
  }

  // ---- stage D tiles (+bias) into LDS ------------------------------------
  {
    const int n0 = nbase + m, n1 = n0 + 16;
    const float b0 = eb[n0], b1 = eb[n1];
    #pragma unroll
    for (int v = 0; v < 8; ++v) {
      const int row = v + 8 * half;
      estage[row * 132 + n0] = acc0[v] + b0;
      estage[row * 132 + n1] = acc1[v] + b1;
    }
  }
  __syncthreads();

  // ---- repack as A-fragments (bf16) for the LSTM x input -----------------
  {
    const int ks = tid >> 5;            // 0..3
    const int L  = tid & 31;
    const int pm = L & 15, ph = L >> 4;
    const int k0 = 32 * ks + 8 * ph;
    V16BF t16;
    #pragma unroll
    for (int i = 0; i < 8; ++i) {
      t16.v[i]     = (__bf16)estage[pm * 132 + k0 + i];
      t16.v[8 + i] = (__bf16)estage[pm * 132 + k0 + 16 + i];
    }
    uint4* dst = (uint4*)embP + (size_t)(((btile * 64 + t) * 4 + ks) * 64 + L * 2);
    dst[0] = t16.q[0];
    dst[1] = t16.q[1];
  }
}

// ---------------------------------------------------------------------------
// Kernel 2: LSTM scan + head. 64 blocks x 512 threads (16 waves).
// Per-wave Wi/Wh fragments register-resident; x_t double-buffered in LDS
// via async global->LDS copies issued one step ahead.
// ---------------------------------------------------------------------------
__global__ __launch_bounds__(512, 1)
void lstm_kernel(const __bf16* __restrict__ embP,
                 const __bf16* __restrict__ WtI,   // [512][128] bf16
                 const __bf16* __restrict__ WtH,   // [512][128] bf16
                 const float* __restrict__ lb,     // [512]
                 const __bf16* __restrict__ WtMu,  // [128][128] bf16
                 const float* __restrict__ mub,    // [128]
                 float* __restrict__ out) {        // mu | log_std
  extern __shared__ char smem[];
  float*  zbuf   = (float*)smem;                       // 16 x 516 f32   (33024 B)
  float*  biasL  = zbuf + 16 * 516;                    // 512 f32        ( 2048 B)
  __bf16* hbuf   = (__bf16*)(biasL + 512);             // 16 x 144 bf16  ( 4608 B)
  char*   xstage = (char*)(hbuf + 16 * 144);           // 2 x 4096 B double buffer

  const int tid  = threadIdx.x;
  const int lane = tid & 31;
  const int wid  = tid >> 5;          // 0..15
  const int m    = lane & 15;
  const int half = lane >> 4;
  const int nb   = wid * 32;
  const int btile = blockIdx.x;       // 16 batch rows

  const char* xg = (const char*)embP + (size_t)btile * 64 * 4096;   // this block's x stream
  const unsigned xLds = (unsigned)(unsigned long long)(void*)xstage;

  // ---- kick off async prefetch of x_0 into buffer 0 -----------------------
  if (wid == 0) async_copy_x4k(xLds, xg, lane);

  // ---- bias to LDS, zero h ------------------------------------------------
  biasL[tid] = lb[tid];
  for (int i = tid; i < 16 * 144 / 2; i += 512) ((unsigned int*)hbuf)[i] = 0u;

  // ---- hoist this wave's Wi/Wh B-fragments into registers -----------------
  v16bf Bi[8], Bh[8];                 // [ks][tile]
  #pragma unroll
  for (int ks = 0; ks < 4; ++ks) {
    const __bf16* pi0 = WtI + (size_t)(nb + m) * 128 + 32 * ks + 8 * half;
    const __bf16* ph0 = WtH + (size_t)(nb + m) * 128 + 32 * ks + 8 * half;
    Bi[2 * ks]     = make_v16bf(*(const uint4*)pi0, *(const uint4*)(pi0 + 16));
    Bi[2 * ks + 1] = make_v16bf(*(const uint4*)(pi0 + 16 * 128), *(const uint4*)(pi0 + 16 * 128 + 16));
    Bh[2 * ks]     = make_v16bf(*(const uint4*)ph0, *(const uint4*)(ph0 + 16));
    Bh[2 * ks + 1] = make_v16bf(*(const uint4*)(ph0 + 16 * 128), *(const uint4*)(ph0 + 16 * 128 + 16));
  }

  if (wid == 0) wait_async0();        // x_0 landed
  __syncthreads();                    // publish x_0 / bias / h=0

  const float bz0 = biasL[nb + m];
  const float bz1 = biasL[nb + 16 + m];

  // elementwise-phase mapping: this thread owns row em, hidden cols ej+32q
  const int em = tid & 15;
  const int ej = tid >> 4;            // 0..31
  float c[4] = {0.f, 0.f, 0.f, 0.f};

  for (int t = 0; t < TT; ++t) {
    // prefetch x_{t+1} into the other buffer (its last readers finished at
    // the end of step t-1, before the barrier we just passed)
    if (wid == 0 && (t + 1) < TT)
      async_copy_x4k(xLds + (((t + 1) & 1) * 4096), xg + (size_t)(t + 1) * 4096, lane);

    const char* xb = xstage + (t & 1) * 4096;
    v8f acc0 = {}; v8f acc1 = {};

    #pragma unroll
    for (int ks = 0; ks < 4; ++ks) {
      // A: x_t fragment from LDS (broadcast to all 16 waves)
      const uint4* xp = (const uint4*)(xb + ks * 1024 + lane * 32);
      v16bf Ax = make_v16bf(xp[0], xp[1]);
      // A: h fragment from LDS
      const __bf16* hp = hbuf + m * 144 + 32 * ks + 8 * half;
      v16bf Ah = make_v16bf(*(const uint4*)hp, *(const uint4*)(hp + 16));

      acc0 = __builtin_amdgcn_wmma_f32_16x16x32_bf16(false, Ax, false, Bi[2 * ks],     (short)0, acc0, false, false);
      acc0 = __builtin_amdgcn_wmma_f32_16x16x32_bf16(false, Ah, false, Bh[2 * ks],     (short)0, acc0, false, false);
      acc1 = __builtin_amdgcn_wmma_f32_16x16x32_bf16(false, Ax, false, Bi[2 * ks + 1], (short)0, acc1, false, false);
      acc1 = __builtin_amdgcn_wmma_f32_16x16x32_bf16(false, Ah, false, Bh[2 * ks + 1], (short)0, acc1, false, false);
    }

    // ---- z tile (+bias) to LDS --------------------------------------------
    #pragma unroll
    for (int v = 0; v < 8; ++v) {
      const int row = v + 8 * half;
      zbuf[row * 516 + nb + m]      = acc0[v] + bz0;
      zbuf[row * 516 + nb + 16 + m] = acc1[v] + bz1;
    }
    __syncthreads();

    // ---- gate activations, c/h update -------------------------------------
    #pragma unroll
    for (int q = 0; q < 4; ++q) {
      const int j = ej + 32 * q;
      const float zi = zbuf[em * 516 + j];
      const float zf = zbuf[em * 516 + 128 + j];
      const float zg = zbuf[em * 516 + 256 + j];
      const float zo = zbuf[em * 516 + 384 + j];
      const float iv = sigmoidf_(zi);
      const float fv = sigmoidf_(zf);
      const float gv = tanhf(zg);
      const float ov = sigmoidf_(zo);
      const float cn = fv * c[q] + iv * gv;
      c[q] = cn;
      hbuf[em * 144 + j] = (__bf16)(ov * tanhf(cn));
    }

    if (wid == 0) wait_async0();      // x_{t+1} landed before publish barrier
    __syncthreads();                  // publish h_t and x_{t+1}
  }

  // ---- head: out = h @ muW + mub; split into mu / clipped log_std ---------
  if (wid < 4) {
    v8f a0 = {}; v8f a1 = {};
    #pragma unroll
    for (int ks = 0; ks < 4; ++ks) {
      const __bf16* hp = hbuf + m * 144 + 32 * ks + 8 * half;
      v16bf Ah = make_v16bf(*(const uint4*)hp, *(const uint4*)(hp + 16));
      const __bf16* mp0 = WtMu + (size_t)(nb + m) * 128 + 32 * ks + 8 * half;
      const __bf16* mp1 = mp0 + 16 * 128;
      v16bf M0 = make_v16bf(*(const uint4*)mp0, *(const uint4*)(mp0 + 16));
      v16bf M1 = make_v16bf(*(const uint4*)mp1, *(const uint4*)(mp1 + 16));
      a0 = __builtin_amdgcn_wmma_f32_16x16x32_bf16(false, Ah, false, M0, (short)0, a0, false, false);
      a1 = __builtin_amdgcn_wmma_f32_16x16x32_bf16(false, Ah, false, M1, (short)0, a1, false, false);
    }
    const int nA = nb + m, nB = nA + 16;
    const float bA = mub[nA], bB = mub[nB];
    #pragma unroll
    for (int v = 0; v < 8; ++v) {
      const int row = btile * 16 + v + 8 * half;
      float vA = a0[v] + bA;
      float vB = a1[v] + bB;
      if (nA < NOUT) out[(size_t)row * NOUT + nA] = vA;
      else out[(size_t)BB * NOUT + (size_t)row * NOUT + (nA - NOUT)] =
               fminf(fmaxf(vA, -10.0f), 2.0f);
      if (nB < NOUT) out[(size_t)row * NOUT + nB] = vB;
      else out[(size_t)BB * NOUT + (size_t)row * NOUT + (nB - NOUT)] =
               fminf(fmaxf(vB, -10.0f), 2.0f);
    }
  }
}

// ---------------------------------------------------------------------------
extern "C" void kernel_launch(void* const* d_in, const int* in_sizes, int n_in,
                              void* d_out, int out_size, void* d_ws, size_t ws_size,
                              hipStream_t stream) {
  const float* obs = (const float*)d_in[0];
  const float* act = (const float*)d_in[1];
  const float* eW  = (const float*)d_in[2];
  const float* eb  = (const float*)d_in[3];
  const float* Wi  = (const float*)d_in[4];
  const float* Wh  = (const float*)d_in[5];
  const float* lb  = (const float*)d_in[6];
  const float* muW = (const float*)d_in[7];
  const float* mub = (const float*)d_in[8];
  float* out = (float*)d_out;

  char* w = (char*)d_ws;
  __bf16* WtE  = (__bf16*)w;  w += (size_t)128 * 288 * 2;   // 73728 B
  __bf16* WtI  = (__bf16*)w;  w += (size_t)512 * 128 * 2;   // 131072 B
  __bf16* WtH  = (__bf16*)w;  w += (size_t)512 * 128 * 2;   // 131072 B
  __bf16* WtMu = (__bf16*)w;  w += (size_t)128 * 128 * 2;   // 32768 B
  __bf16* embP = (__bf16*)w;                                // 16 MiB

  const int prepTot = 128 * 288 + 2 * 512 * 128 + 128 * 128;
  prep_weights<<<(prepTot + 255) / 256, 256, 0, stream>>>(eW, Wi, Wh, muW,
                                                          WtE, WtI, WtH, WtMu);

  embed_kernel<<<dim3(64, 64), 128, 0, stream>>>(obs, act, WtE, eb, embP);

  const size_t smemBytes = (size_t)16 * 516 * 4        // z staging
                         + (size_t)512 * 4             // bias
                         + (size_t)16 * 144 * 2        // h (bf16)
                         + (size_t)2 * 4096;           // x double buffer
  lstm_kernel<<<64, 512, smemBytes, stream>>>(embP, WtI, WtH, lb, WtMu, mub, out);
}
